// SinkhornAttention_71597104824508
// MI455X (gfx1250) — matrix operand order repaired and checked by reading, not compiled, not run on hardware
//
#include <hip/hip_runtime.h>
#include <hip/hip_bf16.h>

typedef __attribute__((ext_vector_type(16))) _Float16 v16h;
typedef __attribute__((ext_vector_type(8)))  _Float16 v8h;
typedef __attribute__((ext_vector_type(2)))  _Float16 v2h;
typedef __attribute__((ext_vector_type(8)))  float    v8f;

#define BH_TOT 32
#define T_LEN  8192
#define DH     64
#define NBUCK  64
#define BSZ    128
#define KV2    256
#define SCL2   (0.04419417382415922f * 1.4426950408889634f)  // 512^-0.5 * log2e
#define LEAKY  0.01f
#define EPSF   1e-6f
#define INV_TEMP (1.0f/0.75f)
#define VT_STRIDE 264                 // 256 + 8 pad: 528B rows, 16B-aligned

// ---------------------------------------------------------------------------
// Kernel 1: bucket sums -> SortNet -> Gumbel-Sinkhorn -> top-1 (index, weight)
// ---------------------------------------------------------------------------
__global__ __launch_bounds__(256) void sortnet_sinkhorn_kernel(
    const float* __restrict__ q, const float* __restrict__ k,
    const float* __restrict__ W, int* __restrict__ vsel, float* __restrict__ wsel)
{
    __shared__ float xs[NBUCK][128];   // [bucket][2*dh] summary
    __shared__ float Rm[NBUCK][65];    // padded to kill bank conflicts
    const int bh  = blockIdx.x;
    const int hh  = bh & 7;            // head index (bh = b*8 + h)
    const int tid = threadIdx.x;
    const size_t base = (size_t)bh * T_LEN * DH;

    for (int s = tid; s < NBUCK * 128; s += 256) {
        const int e = s & 127;
        const int u = s >> 7;
        const float* src = (e < 64) ? (q + base + (size_t)u * BSZ * DH + e)
                                    : (k + base + (size_t)u * BSZ * DH + (e - 64));
        float acc = 0.f;
        for (int r = 0; r < BSZ; ++r) acc += src[r * DH];
        xs[u][e] = acc;
    }
    __syncthreads();

    for (int s = tid; s < NBUCK * NBUCK; s += 256) {
        const int o = s & 63;
        const int u = s >> 6;
        const float* wp = W + (size_t)hh * 128 * 64 + o;
        float acc = 0.f;
        for (int e = 0; e < 128; ++e) acc += xs[u][e] * wp[e * 64];
        float R = acc > 0.f ? acc : LEAKY * acc;
        unsigned hsh = (unsigned)(bh * 4096 + s) * 2654435769u + 42u;
        hsh ^= hsh >> 16; hsh *= 2246822519u;
        hsh ^= hsh >> 13; hsh *= 3266489917u;
        hsh ^= hsh >> 16;
        float uu = (float)(hsh >> 8) * (1.0f / 16777216.0f);
        float g  = -__logf(-__logf(uu + EPSF) + EPSF);
        Rm[u][o] = (__logf(R + EPSF) + g) * INV_TEMP;
    }
    __syncthreads();

    for (int it = 0; it < 7; ++it) {
        if (tid < 64) {
            float mx = -1e30f;
            for (int o = 0; o < 64; ++o) mx = fmaxf(mx, Rm[tid][o]);
            float sm = 0.f;
            for (int o = 0; o < 64; ++o) sm += __expf(Rm[tid][o] - mx);
            float lse = mx + __logf(sm);
            for (int o = 0; o < 64; ++o) Rm[tid][o] -= lse;
        }
        __syncthreads();
        if (tid < 64) {
            float mx = -1e30f;
            for (int u = 0; u < 64; ++u) mx = fmaxf(mx, Rm[u][tid]);
            float sm = 0.f;
            for (int u = 0; u < 64; ++u) sm += __expf(Rm[u][tid] - mx);
            float lse = mx + __logf(sm);
            for (int u = 0; u < 64; ++u) Rm[u][tid] -= lse;
        }
        __syncthreads();
    }

    if (tid < 64) {
        float best = -1e30f; int bi = 0;
        for (int o = 0; o < 64; ++o) {
            float x = Rm[tid][o];
            if (x > best) { best = x; bi = o; }
        }
        vsel[bh * 64 + tid] = bi;
        wsel[bh * 64 + tid] = __expf(best);
    }
}

// ---------------------------------------------------------------------------
// Kernel 2: per-(slab,bucket) attention with f16 WMMA + online softmax
// ---------------------------------------------------------------------------
__device__ __forceinline__ v16h ld16h(const _Float16* p) {
    v8h lo = *(const v8h*)p;
    v8h hi = *(const v8h*)(p + 8);
    v16h r;
#pragma unroll
    for (int i = 0; i < 8; ++i) { r[i] = lo[i]; r[i + 8] = hi[i]; }
    return r;
}

// Single-instruction DPP-fused reduction steps: dst = op(dpp(x), x).
// Butterfly stays inside each 16-lane half (quad swaps, half-mirror, mirror),
// matching the C/D-matrix row split. VALU-only; no canonicalize, no LDS.
#define DEF_DPP_STEP(NAME, INSTR, CTRL)                                      \
__device__ __forceinline__ float NAME(float x) {                             \
    float r;                                                                 \
    asm(INSTR " %0, %1, %2 " CTRL " row_mask:0xf bank_mask:0xf bound_ctrl:1" \
        : "=v"(r) : "v"(x), "v"(x));                                         \
    return r;                                                                \
}
DEF_DPP_STEP(max_q1, "v_max_num_f32_dpp", "quad_perm:[1,0,3,2]")
DEF_DPP_STEP(max_q2, "v_max_num_f32_dpp", "quad_perm:[2,3,0,1]")
DEF_DPP_STEP(max_hm, "v_max_num_f32_dpp", "row_half_mirror")
DEF_DPP_STEP(max_mr, "v_max_num_f32_dpp", "row_mirror")
DEF_DPP_STEP(add_q1, "v_add_f32_dpp",     "quad_perm:[1,0,3,2]")
DEF_DPP_STEP(add_q2, "v_add_f32_dpp",     "quad_perm:[2,3,0,1]")
DEF_DPP_STEP(add_hm, "v_add_f32_dpp",     "row_half_mirror")
DEF_DPP_STEP(add_mr, "v_add_f32_dpp",     "row_mirror")

__device__ __forceinline__ float redmax16(float x) {
    return max_mr(max_hm(max_q2(max_q1(x))));
}
__device__ __forceinline__ float redsum16(float x) {
    return add_mr(add_hm(add_q2(add_q1(x))));
}
// plain max without LLVM's canonicalize pairs
__device__ __forceinline__ float vmaxf(float a, float b) {
    float r;
    asm("v_max_num_f32 %0, %1, %2" : "=v"(r) : "v"(a), "v"(b));
    return r;
}

__global__ __launch_bounds__(256) void attn_kernel(
    const float* __restrict__ q, const float* __restrict__ k,
    const float* __restrict__ v, const int* __restrict__ vsel,
    const float* __restrict__ wsel, float* __restrict__ out)
{
    __shared__ _Float16 Ks[KV2][DH];            // 32 KB: [w*K[v*] ; K[u]] row-major
    __shared__ _Float16 Vt[DH][VT_STRIDE];      // ~33 KB: transposed V tile
    __shared__ _Float16 Ps[8][16][32];          // 8 KB: per-wave P staging

    const int blk  = blockIdx.x;
    const int bh   = blk >> 6;
    const int u    = blk & 63;
    const int tid  = threadIdx.x;
    const int lane = tid & 31;
    const int wid  = tid >> 5;
    const int m    = lane & 15;
    const int hi   = lane >> 4;                 // 0 | 1 (16-lane half)

    const int   vs = vsel[bh * 64 + u];
    const float w  = wsel[bh * 64 + u];

    const size_t base = (size_t)bh * T_LEN * DH;
    const float* kg_r = k + base + (size_t)vs * BSZ * DH;
    const float* kg_u = k + base + (size_t)u  * BSZ * DH;
    const float* vg_r = v + base + (size_t)vs * BSZ * DH;
    const float* vg_u = v + base + (size_t)u  * BSZ * DH;

    // cooperative fill: paired f32 loads, packed f16 stores for K; V transposed
    for (int idx = tid; idx < (BSZ * DH) / 2; idx += 256) {
        const int r  = idx >> 5;
        const int c2 = (idx & 31) << 1;
        const float2 a = *(const float2*)(kg_r + r * DH + c2);
        const float2 b = *(const float2*)(kg_u + r * DH + c2);
        *(v2h*)&Ks[r][c2]       = (v2h){(_Float16)(a.x * w), (_Float16)(a.y * w)};
        *(v2h*)&Ks[BSZ + r][c2] = (v2h){(_Float16)b.x, (_Float16)b.y};
        const float2 c = *(const float2*)(vg_r + r * DH + c2);
        const float2 d = *(const float2*)(vg_u + r * DH + c2);
        Vt[c2][r]           = (_Float16)(c.x * w);
        Vt[c2 + 1][r]       = (_Float16)(c.y * w);
        Vt[c2][BSZ + r]     = (_Float16)d.x;
        Vt[c2 + 1][BSZ + r] = (_Float16)d.y;
    }

    // Q strip A-fragments (16-bit A layout), pre-scaled by 512^-0.5 * log2e
    // so the whole softmax runs in the exp2 domain (v_exp_f32 is native exp2).
    const float* qg = q + base + (size_t)u * BSZ * DH + (size_t)(wid * 16) * DH;
    v16h aQ[2];
#pragma unroll
    for (int f = 0; f < 2; ++f) {
#pragma unroll
        for (int j = 0; j < 16; j += 2) {
            const int kk = f * 32 + hi * 16 + j;
            const float2 val = *(const float2*)(qg + m * DH + kk);
            aQ[f][j]     = (_Float16)(val.x * SCL2);
            aQ[f][j + 1] = (_Float16)(val.y * SCL2);
        }
    }
    __syncthreads();

    float mrow[8], lrow[8];
#pragma unroll
    for (int r = 0; r < 8; ++r) { mrow[r] = -INFINITY; lrow[r] = 0.f; }
    v8f Oacc[4] = {};

    // online softmax over 8 chunks of 32 K/V columns
    for (int c = 0; c < 8; ++c) {
        v8f s[2];
#pragma unroll
        for (int tt = 0; tt < 2; ++tt) {
            const int n0 = c * 32 + tt * 16;
            v8f acc = {};
#pragma unroll
            for (int f = 0; f < 2; ++f) {
                v16h bfrag = ld16h(&Ks[n0 + m][f * 32 + hi * 16]);
                acc = __builtin_amdgcn_wmma_f32_16x16x32_f16(
                          false, aQ[f], false, bfrag, (short)0, acc, false, false);
            }
            s[tt] = acc;
        }

        // per-row running max (fused DPP butterflies) and rescale factor
        float alpha[8];
#pragma unroll
        for (int r = 0; r < 8; ++r) {
            const float cm = redmax16(vmaxf(s[0][r], s[1][r]));
            const float mn = vmaxf(mrow[r], cm);
            alpha[r] = __builtin_amdgcn_exp2f(mrow[r] - mn);
            mrow[r]  = mn;
        }
        // p = exp2(s - m); denominator kept as PER-LANE partials: alpha is
        // uniform across a row's 16 lanes so partials rescale consistently;
        // one cross-lane sum after the loop suffices.
#pragma unroll
        for (int r = 0; r < 8; ++r) {
            const float p0 = __builtin_amdgcn_exp2f(s[0][r] - mrow[r]);
            const float p1 = __builtin_amdgcn_exp2f(s[1][r] - mrow[r]);
            s[0][r] = p0; s[1][r] = p1;
            lrow[r] = lrow[r] * alpha[r] + (p0 + p1);
        }
#pragma unroll
        for (int t4 = 0; t4 < 4; ++t4)
#pragma unroll
            for (int r = 0; r < 8; ++r) Oacc[t4][r] *= alpha[r];

        // stage P (D layout -> row-major) through per-wave LDS, re-read as A
#pragma unroll
        for (int tt = 0; tt < 2; ++tt)
#pragma unroll
            for (int r = 0; r < 8; ++r)
                Ps[wid][r + hi * 8][m + tt * 16] = (_Float16)s[tt][r];
        asm volatile("s_wait_dscnt 0" ::: "memory");

        v16h aP;
#pragma unroll
        for (int j = 0; j < 16; j += 2) {
            const int kk = hi * 16 + j;
            aP[j]     = Ps[wid][m][kk];
            aP[j + 1] = Ps[wid][m][kk + 1];
        }

#pragma unroll
        for (int t4 = 0; t4 < 4; ++t4) {
            v16h bv = ld16h(&Vt[t4 * 16 + m][c * 32 + hi * 16]);
            Oacc[t4] = __builtin_amdgcn_wmma_f32_16x16x32_f16(
                           false, aP, false, bv, (short)0, Oacc[t4], false, false);
        }
    }

    // finalize: one cross-lane sum for the denominator, then store
    float inv[8];
#pragma unroll
    for (int r = 0; r < 8; ++r) inv[r] = 1.0f / redsum16(lrow[r]);

    float* og = out + base + (size_t)u * BSZ * DH + (size_t)(wid * 16) * DH;
#pragma unroll
    for (int t4 = 0; t4 < 4; ++t4) {
#pragma unroll
        for (int r = 0; r < 8; ++r) {
            const int row = r + hi * 8;
            og[row * DH + t4 * 16 + m] = Oacc[t4][r] * inv[r];
        }
    }
}

// ---------------------------------------------------------------------------
extern "C" void kernel_launch(void* const* d_in, const int* in_sizes, int n_in,
                              void* d_out, int out_size, void* d_ws, size_t ws_size,
                              hipStream_t stream) {
    const float* q = (const float*)d_in[0];
    const float* k = (const float*)d_in[1];
    const float* v = (const float*)d_in[2];
    const float* W = (const float*)d_in[3];
    float* out = (float*)d_out;

    int*   vsel = (int*)d_ws;
    float* wsel = (float*)((char*)d_ws + (size_t)BH_TOT * NBUCK * sizeof(int));

    sortnet_sinkhorn_kernel<<<BH_TOT, 256, 0, stream>>>(q, k, W, vsel, wsel);
    attn_kernel<<<BH_TOT * NBUCK, 256, 0, stream>>>(q, k, v, vsel, wsel, out);
}
